// FPNPostProcessor_25975962206917
// MI455X (gfx1250) — compile-verified, matrix operand order checked by reading, not run on hardware
//
#include <hip/hip_runtime.h>
#include <hip/hip_bf16.h>
#include <math.h>

// ---------------- problem constants (must match reference) ----------------
#define BB        2          // images
#define NN        1000       // proposals / image
#define CC        81         // classes incl. background
#define CM1       80         // foreground classes
#define KK        100        // detections / image
#define NPAD      1024       // proposals padded to pow2 for bitonic sort
#define SCORE_TH  0.05f
#define NMS_TH    0.5f
#define IMG_W     1216.0f
#define IMG_H     800.0f
#define XFORM_CLIP 4.135166556742356f   // log(1000/16)
#define AREA_SCALE (1.0f/1024.0f)       // keep areas inside fp16 range
#define NEG_INF   (-3.0e38f)

typedef __attribute__((ext_vector_type(16))) _Float16 v16h;
typedef __attribute__((ext_vector_type(8)))  float    v8f;
typedef __attribute__((ext_vector_type(4)))  unsigned uint32x4;
typedef __attribute__((ext_vector_type(8)))  int      int32x8;
typedef __attribute__((ext_vector_type(4)))  int      int32x4;

// LDS carve offsets (bytes) for kernel 2 — dynamic LDS assumed to start at 0
#define LDS_WORDS_CORE  (NPAD * 7 + NPAD * 32 + 32)          // s/idx/box/area/sup/keep
#define STAGE_OFF_BYTES ((unsigned)(LDS_WORDS_CORE * 4))     // 159,872 (16B aligned)
#define STAGE_ELEMS     (NN * 4)                             // 4000 floats = 16,000 B

// =====================================================================
// Kernel 1: per-proposal softmax + box decode + clip.
// grid = B*N blocks, 128 threads. Writes class-major scores [B,CM1,N]
// and boxes [B,CM1,N,4].
// =====================================================================
__global__ void fpn_softmax_decode(const float* __restrict__ logits,
                                   const float* __restrict__ boxreg,
                                   const float* __restrict__ rois,
                                   float* __restrict__ sc_out,
                                   float* __restrict__ bx_out)
{
    const int row = blockIdx.x;            // b*N + n
    const int b   = row / NN;
    const int n   = row - b * NN;
    const int tid = threadIdx.x;

    __shared__ float red[128];

    float l = (tid < CC) ? logits[(size_t)row * CC + tid] : NEG_INF;

    red[tid] = l; __syncthreads();
    for (int off = 64; off > 0; off >>= 1) {
        if (tid < off) red[tid] = fmaxf(red[tid], red[tid + off]);
        __syncthreads();
    }
    const float mx = red[0];
    __syncthreads();

    const float e = (tid < CC) ? __expf(l - mx) : 0.0f;
    red[tid] = e; __syncthreads();
    for (int off = 64; off > 0; off >>= 1) {
        if (tid < off) red[tid] += red[tid + off];
        __syncthreads();
    }
    const float denom = red[0];

    if (tid >= 1 && tid < CC) {
        const int c = tid;
        const float prob = e / denom;

        const float* rp = &rois[(size_t)row * 4];
        const float x1 = rp[0], y1 = rp[1], x2 = rp[2], y2 = rp[3];
        const float w  = x2 - x1 + 1.0f;
        const float h  = y2 - y1 + 1.0f;
        const float cx = x1 + 0.5f * w;
        const float cy = y1 + 0.5f * h;

        const float* rc = &boxreg[(size_t)row * 4 * CC + 4 * c];
        const float dx = rc[0] * 0.1f;
        const float dy = rc[1] * 0.1f;
        const float dw = fminf(rc[2] * 0.2f, XFORM_CLIP);
        const float dh = fminf(rc[3] * 0.2f, XFORM_CLIP);

        const float pcx = dx * w + cx;
        const float pcy = dy * h + cy;
        const float pw  = __expf(dw) * w;
        const float ph  = __expf(dh) * h;

        float bx1 = pcx - 0.5f * pw;
        float by1 = pcy - 0.5f * ph;
        float bx2 = pcx + 0.5f * pw - 1.0f;
        float by2 = pcy + 0.5f * ph - 1.0f;
        bx1 = fminf(fmaxf(bx1, 0.0f), IMG_W - 1.0f);
        by1 = fminf(fmaxf(by1, 0.0f), IMG_H - 1.0f);
        bx2 = fminf(fmaxf(bx2, 0.0f), IMG_W - 1.0f);
        by2 = fminf(fmaxf(by2, 0.0f), IMG_H - 1.0f);

        const size_t o = ((size_t)b * CM1 + (c - 1)) * NN + n;
        sc_out[o]       = prob;
        bx_out[o*4 + 0] = bx1;
        bx_out[o*4 + 1] = by1;
        bx_out[o*4 + 2] = bx2;
        bx_out[o*4 + 3] = by2;
    }
}

// ---------------------------------------------------------------------
// Issue one TDM tensor_load_to_lds: contiguous 1-D tile of NN*4 f32
// (this class's boxes) from global -> LDS staging buffer. Descriptor
// layout per CDNA5 ISA ch.8 (D# groups 0/1; groups 2/3 unused for 1-D).
// ---------------------------------------------------------------------
__device__ __forceinline__ void tdm_stage_boxes(const float* gsrc)
{
    const unsigned long long ga = (unsigned long long)(const void*)gsrc;

    uint32x4 g0 = {};
    g0[0] = 1u;                                  // count=1, is_restore=0, no gather
    g0[1] = STAGE_OFF_BYTES;                     // lds_addr
    g0[2] = (unsigned)(ga & 0xffffffffull);      // global_addr[31:0]
    g0[3] = (unsigned)((ga >> 32) & 0x01ffffffull) | (2u << 30); // addr[56:32], type=2

    int32x8 g1 = {};
    g1[0] = (int)(2u << 16);                     // wg_mask=0, data_size=2 (4B)
    g1[1] = (int)((STAGE_ELEMS & 0xffff) << 16); // tensor_dim0[15:0] @ bits 63:48
    g1[2] = (int)(STAGE_ELEMS >> 16);            // tensor_dim0[31:16]; tensor_dim1=0
    g1[3] = (int)((STAGE_ELEMS & 0xffff) << 16); // tile_dim0 @ bits 127:112
    g1[4] = 0;                                   // tile_dim1=0, tile_dim2=0 (1-D)
    g1[5] = (int)STAGE_ELEMS;                    // tensor_dim0_stride[31:0]
    g1[6] = 0;
    g1[7] = 0;

    int32x4 g2 = {};                             // unused (<=2-D tensor)
    int32x4 g3 = {};

#if __clang_major__ >= 23
    int32x8 g4 = {};
    __builtin_amdgcn_tensor_load_to_lds(g0, g1, g2, g3, g4, 0);
#else
    __builtin_amdgcn_tensor_load_to_lds(g0, g1, g2, g3, 0);
#endif
}

// =====================================================================
// Kernel 2: per-(image,class) sort + NMS. grid = B*CM1 = 160 blocks,
// 256 threads (8 waves). ~176KB LDS (CDNA5: 320KB/WGP):
//   scores/idx/boxes/areas + 1024x1024-bit suppression matrix (128KB)
//   + 16KB TDM staging buffer for this class's raw boxes.
// Pipeline: TDM DMA of boxes overlaps the bitonic sort; pairwise
// area sums via v_wmma_f32_16x16x32_f16 co-execute with VALU min/max
// intersection math; serial greedy chain runs on one wave32 w/ shuffles.
// =====================================================================
__global__ void fpn_sort_nms(const float* __restrict__ sc_in,
                             const float* __restrict__ bx_in,
                             float* __restrict__ sc_msk,
                             float* __restrict__ bx_srt)
{
    extern __shared__ unsigned char smem[];
    float*    sh_s    = (float*)smem;                         // [1024]
    int*      sh_idx  = (int*)(sh_s + NPAD);                  // [1024]
    float*    sh_x1   = (float*)(sh_idx + NPAD);              // [1024]
    float*    sh_y1   = sh_x1 + NPAD;
    float*    sh_x2   = sh_y1 + NPAD;
    float*    sh_y2   = sh_x2 + NPAD;
    float*    sh_area = sh_y2 + NPAD;                         // scaled by 2^-10
    unsigned* sh_sup  = (unsigned*)(sh_area + NPAD);          // [1024][32]
    unsigned* sh_keep = sh_sup + NPAD * 32;                   // [32]
    float*    sh_stage= (float*)(sh_keep + 32);               // [NN*4] raw boxes

    const int bc   = blockIdx.x;       // b*CM1 + c
    const int tid  = threadIdx.x;
    const int lane = tid & 31;
    const int wave = tid >> 5;

    // ---- kick off TDM DMA of this class's boxes (overlaps the sort) ----
    if (tid == 0) {
        tdm_stage_boxes(&bx_in[(size_t)bc * NN * 4]);
    }
    __builtin_prefetch(&sc_in[(size_t)bc * NN], 0, 1);  // global_prefetch scores

    // ---- load scores, pad to 1024 ----
    for (int m = tid; m < NPAD; m += 256) {
        sh_s[m]   = (m < NN) ? sc_in[(size_t)bc * NN + m] : NEG_INF;
        sh_idx[m] = m;
    }
    __syncthreads();

    // ---- bitonic sort, descending, carry original index ----
    for (int kk = 2; kk <= NPAD; kk <<= 1) {
        for (int jj = kk >> 1; jj > 0; jj >>= 1) {
            for (int m = tid; m < NPAD; m += 256) {
                const int p = m ^ jj;
                if (p > m) {
                    const bool desc = ((m & kk) == 0);
                    const float sm = sh_s[m], sp = sh_s[p];
                    const bool sw = desc ? (sm < sp) : (sm > sp);
                    if (sw) {
                        sh_s[m] = sp; sh_s[p] = sm;
                        const int t = sh_idx[m]; sh_idx[m] = sh_idx[p]; sh_idx[p] = t;
                    }
                }
            }
            __syncthreads();
        }
    }

    // ---- TDM complete? (TENSORcnt nonzero only on issuing wave) ----
    __builtin_amdgcn_s_wait_tensorcnt(0);
    __syncthreads();

    // ---- gather boxes in sorted order from LDS staging ----
    for (int m = tid; m < NPAD; m += 256) {
        const int o = sh_idx[m];
        float x1 = 0.f, y1 = 0.f, x2 = 0.f, y2 = 0.f;
        if (o < NN) {
            const float* p = &sh_stage[o * 4];
            x1 = p[0]; y1 = p[1]; x2 = p[2]; y2 = p[3];
        }
        sh_x1[m] = x1; sh_y1[m] = y1; sh_x2[m] = x2; sh_y2[m] = y2;
        sh_area[m] = (x2 - x1 + 1.0f) * (y2 - y1 + 1.0f) * AREA_SCALE;
    }
    for (int m = tid; m < NPAD * 32; m += 256) sh_sup[m] = 0u;
    __syncthreads();

    // ---- pairwise IoU -> suppression bitmask, 16x16 tiles per wave ----
    // union denominator (area_i + area_j) via WMMA rank-2 product:
    //   A[M][0]=area_i[M], A[M][1]=1 ; B[0][N]=1, B[1][N]=area_j[N]
    for (int t = wave; t < 64 * 64; t += 8) {
        const int ti = t >> 6, tj = t & 63;
        const int m16 = lane & 15;
        const bool lo = (lane < 16);

        v16h a = {};   // A 16x32 f16: lanes 0-15 hold K=0..7 in halves 0..7
        v16h bm = {};  // B 32x16 f16: lanes 0-15 hold K=0..7 in halves 0..7
        a[0]  = (_Float16)(lo ? sh_area[ti * 16 + m16] : 0.0f);
        a[1]  = lo ? (_Float16)1.0f : (_Float16)0.0f;
        bm[0] = lo ? (_Float16)1.0f : (_Float16)0.0f;
        bm[1] = (_Float16)(lo ? sh_area[tj * 16 + m16] : 0.0f);

        v8f acc = {};
        acc = __builtin_amdgcn_wmma_f32_16x16x32_f16(
            false, a, false, bm, (short)0, acc, false, false);

        // D layout: vgpr r, lanes 0-15 -> M=r ; lanes 16-31 -> M=r+8 ; N=lane%16
        const int j = tj * 16 + m16;
        const float jx1 = sh_x1[j], jy1 = sh_y1[j];
        const float jx2 = sh_x2[j], jy2 = sh_y2[j];
        const int rbase = ti * 16 + ((lane >> 4) << 3);

        #pragma unroll
        for (int r = 0; r < 8; ++r) {
            const int i = rbase + r;
            const float xx1 = fmaxf(sh_x1[i], jx1);
            const float yy1 = fmaxf(sh_y1[i], jy1);
            const float xx2 = fminf(sh_x2[i], jx2);
            const float yy2 = fminf(sh_y2[i], jy2);
            const float w = fmaxf(xx2 - xx1 + 1.0f, 0.0f);
            const float h = fmaxf(yy2 - yy1 + 1.0f, 0.0f);
            const float interS = w * h * AREA_SCALE;
            const float unionS = acc[r] - interS;          // a_i + a_j - inter
            if (interS > NMS_TH * unionS) {                // iou > thresh, no div
                atomicOr(&sh_sup[i * 32 + (j >> 5)], 1u << (j & 31));
            }
        }
    }
    __syncthreads();

    // ---- greedy NMS: single wave32, suppression mask in registers ----
    if (tid < 32) {
        unsigned suppAcc = 0u;     // lane w owns word w of the 1024-bit mask
        unsigned keepM   = 0u;     // lane w owns keep bits [32w, 32w+31]
        for (int i = 0; i < NPAD; ++i) {
            const unsigned wsel = (unsigned)__shfl((int)suppAcc, i >> 5, 32);
            const bool suppressed = (wsel >> (i & 31)) & 1u;
            const bool ki = (sh_s[i] > SCORE_TH) && !suppressed;  // uniform
            if (ki) {
                suppAcc |= sh_sup[i * 32 + tid];
                if (tid == (i >> 5)) keepM |= (1u << (i & 31));
            }
        }
        sh_keep[tid] = keepM;
    }
    __syncthreads();

    // ---- emit masked sorted scores + sorted boxes ----
    for (int m = tid; m < NN; m += 256) {
        const bool keep = (sh_keep[m >> 5] >> (m & 31)) & 1u;
        sc_msk[(size_t)bc * NN + m] = keep ? sh_s[m] : -1.0f;
        float* op = &bx_srt[((size_t)bc * NN + m) * 4];
        op[0] = sh_x1[m]; op[1] = sh_y1[m]; op[2] = sh_x2[m]; op[3] = sh_y2[m];
    }
}

// =====================================================================
// Kernel 3: per-image top-K over 80*1000 masked scores (L2-resident).
// =====================================================================
__global__ void fpn_topk(float* __restrict__ sc_msk,
                         const float* __restrict__ bx_srt,
                         float* __restrict__ out)
{
    const int b   = blockIdx.x;
    const int tid = threadIdx.x;
    const int M   = CM1 * NN;

    __shared__ float rs[256];
    __shared__ int   ri[256];

    for (int k = 0; k < KK; ++k) {
        float best = NEG_INF; int bi = 0;
        for (int m = tid; m < M; m += 256) {
            const float v = sc_msk[(size_t)b * M + m];
            if (v > best || (v == best && m < bi)) { best = v; bi = m; }
        }
        rs[tid] = best; ri[tid] = bi;
        __syncthreads();
        for (int off = 128; off > 0; off >>= 1) {
            if (tid < off) {
                const float ov = rs[tid + off];
                const int   oi = ri[tid + off];
                if (ov > rs[tid] || (ov == rs[tid] && oi < ri[tid])) {
                    rs[tid] = ov; ri[tid] = oi;
                }
            }
            __syncthreads();
        }
        if (tid == 0) {
            const float bs = rs[0];
            const int flat = ri[0];
            const int c = flat / NN;
            const int j = flat - c * NN;
            out[b * KK + k] = bs;
            const float* bp = &bx_srt[(((size_t)b * CM1 + c) * NN + j) * 4];
            float* ob = &out[BB * KK + ((size_t)b * KK + k) * 4];
            ob[0] = bp[0]; ob[1] = bp[1]; ob[2] = bp[2]; ob[3] = bp[3];
            out[BB * KK * 5 + b * KK + k] = (float)(c + 1);
            sc_msk[(size_t)b * M + flat] = NEG_INF;   // exclude from next pick
        }
        __syncthreads();
    }
}

// =====================================================================
extern "C" void kernel_launch(void* const* d_in, const int* in_sizes, int n_in,
                              void* d_out, int out_size, void* d_ws, size_t ws_size,
                              hipStream_t stream)
{
    (void)in_sizes; (void)n_in; (void)out_size; (void)ws_size;

    const float* logits = (const float*)d_in[0];   // [B,N,C]
    const float* boxreg = (const float*)d_in[1];   // [B,N,4C]
    const float* rois   = (const float*)d_in[2];   // [B,N,4]

    float* ws     = (float*)d_ws;
    float* sc_raw = ws;                                  // B*CM1*N
    float* bx_raw = sc_raw + (size_t)BB * CM1 * NN;      // B*CM1*N*4
    float* sc_msk = bx_raw + (size_t)BB * CM1 * NN * 4;  // B*CM1*N
    float* bx_srt = sc_msk + (size_t)BB * CM1 * NN;      // B*CM1*N*4

    float* out = (float*)d_out;

    // 1) softmax + decode
    fpn_softmax_decode<<<BB * NN, 128, 0, stream>>>(logits, boxreg, rois,
                                                    sc_raw, bx_raw);

    // 2) per-(image,class) sort + NMS; core LDS + TDM staging buffer
    const size_t smem2 = (size_t)STAGE_OFF_BYTES + (size_t)STAGE_ELEMS * sizeof(float);
    fpn_sort_nms<<<BB * CM1, 256, smem2, stream>>>(sc_raw, bx_raw,
                                                   sc_msk, bx_srt);

    // 3) per-image top-K
    fpn_topk<<<BB, 256, 0, stream>>>(sc_msk, bx_srt, out);
}
